// BiMambaBlock_72816875536911
// MI455X (gfx1250) — compile-verified, matrix operand order checked by reading, not run on hardware
//
#include <hip/hip_runtime.h>
#include <math.h>

// ---------------------------------------------------------------------------
// BiMamba block for MI455X (gfx1250, wave32, WMMA bf16 16x16x32)
// ---------------------------------------------------------------------------
#define DIMC   256
#define D_IN   128
#define D_STATE 16
#define DT_RANK 16
#define SEQ    1024      // t = L / F
#define NSEQ   24        // b * F
#define MROWS  (NSEQ*SEQ)   // 24576 rows, flat view of all reshapes
#define HID    1024      // EXP * DIM

typedef __attribute__((ext_vector_type(16))) __bf16 v16bf;
typedef __attribute__((ext_vector_type(8)))  float  v8f;

// Load one WMMA operand slice for this lane: p = rowStart + k0 + hi*8.
// Per ISA 16-bit A layout: elems 0..7 <- p[0..7], elems 8..15 <- p[16..23].
__device__ __forceinline__ v16bf ld_op_bf16(const float* p) {
    const float4* q = (const float4*)p;
    float4 c0 = q[0], c1 = q[1], c2 = q[4], c3 = q[5];
    v16bf v;
    v[0]=(__bf16)c0.x; v[1]=(__bf16)c0.y; v[2]=(__bf16)c0.z; v[3]=(__bf16)c0.w;
    v[4]=(__bf16)c1.x; v[5]=(__bf16)c1.y; v[6]=(__bf16)c1.z; v[7]=(__bf16)c1.w;
    v[8]=(__bf16)c2.x; v[9]=(__bf16)c2.y; v[10]=(__bf16)c2.z; v[11]=(__bf16)c2.w;
    v[12]=(__bf16)c3.x; v[13]=(__bf16)c3.y; v[14]=(__bf16)c3.z; v[15]=(__bf16)c3.w;
    return v;
}

#define WMMA_BF16(a, b, c) \
    __builtin_amdgcn_wmma_f32_16x16x32_bf16(false, (a), false, (b), (short)0, (c), false, false)

// out[M,N] = A[M,K] * W[N,K]^T + bias + (resid ? resid : 0)
// Each wave computes a 32x32 output block (2x2 grid of 16x16 WMMA tiles).
// grid: (M/128, N/32), block: 128 (4 waves)
__global__ void gemm_wmma_kernel(const float* __restrict__ A,
                                 const float* __restrict__ W,
                                 const float* __restrict__ bias,
                                 const float* __restrict__ resid,
                                 float* __restrict__ out,
                                 int N, int K) {
    const int lane   = threadIdx.x & 31;
    const int wid    = threadIdx.x >> 5;
    const int mtile0 = blockIdx.x * 8 + wid * 2;   // 2 consecutive M tiles / wave
    const int ntile0 = blockIdx.y * 2;             // 2 consecutive N tiles / wave
    const int row    = lane & 15;
    const int hi     = lane >> 4;

    const float* a0 = A + (size_t)(mtile0 * 16 + row) * K + hi * 8;
    const float* a1 = a0 + (size_t)16 * K;
    const float* w0 = W + (size_t)(ntile0 * 16 + row) * K + hi * 8;
    const float* w1 = w0 + (size_t)16 * K;

    v8f acc00 = {}, acc01 = {}, acc10 = {}, acc11 = {};
    for (int k0 = 0; k0 < K; k0 += 32) {
        __builtin_prefetch(a0 + k0 + 64, 0, 1);    // global_prefetch_b8
        __builtin_prefetch(a1 + k0 + 64, 0, 1);
        v16bf av0 = ld_op_bf16(a0 + k0);
        v16bf av1 = ld_op_bf16(a1 + k0);
        v16bf bv0 = ld_op_bf16(w0 + k0);
        v16bf bv1 = ld_op_bf16(w1 + k0);
        acc00 = WMMA_BF16(av0, bv0, acc00);
        acc01 = WMMA_BF16(av0, bv1, acc01);
        acc10 = WMMA_BF16(av1, bv0, acc10);
        acc11 = WMMA_BF16(av1, bv1, acc11);
    }

    // epilogue: C/D layout -> m = mtile*16 + e + hi*8, n = ntile*16 + (lane&15)
    const int col0 = ntile0 * 16 + row;
    const int col1 = col0 + 16;
    const float b0 = bias ? bias[col0] : 0.f;
    const float b1 = bias ? bias[col1] : 0.f;
    const int m0 = mtile0 * 16 + hi * 8;
    const int m1 = m0 + 16;

    if (resid) {
        #pragma unroll
        for (int e = 0; e < 8; ++e) {
            size_t o00 = (size_t)(m0 + e) * N + col0;
            size_t o01 = (size_t)(m0 + e) * N + col1;
            size_t o10 = (size_t)(m1 + e) * N + col0;
            size_t o11 = (size_t)(m1 + e) * N + col1;
            out[o00] = acc00[e] + b0 + resid[o00];
            out[o01] = acc01[e] + b1 + resid[o01];
            out[o10] = acc10[e] + b0 + resid[o10];
            out[o11] = acc11[e] + b1 + resid[o11];
        }
    } else {
        #pragma unroll
        for (int e = 0; e < 8; ++e) {
            size_t r0 = (size_t)(m0 + e) * N;
            size_t r1 = (size_t)(m1 + e) * N;
            out[r0 + col0] = acc00[e] + b0;
            out[r0 + col1] = acc01[e] + b1;
            out[r1 + col0] = acc10[e] + b0;
            out[r1 + col1] = acc11[e] + b1;
        }
    }
}

// depthwise conv (groups=128, in=2ch/group, taps=3) along t + SiLU
// grid: MROWS blocks, 128 threads
__global__ void conv_silu_kernel(const float* __restrict__ h1,
                                 const float* __restrict__ cw,  // [128][2][3]
                                 const float* __restrict__ cb,  // [128]
                                 float* __restrict__ u) {
    int rowi = blockIdx.x;
    int d = threadIdx.x;
    int b = rowi >> 10, l = rowi & 1023;
    float s = cb[d];
    #pragma unroll
    for (int j = 0; j < 3; ++j) {
        int ll = l + j - 1;
        if (ll >= 0 && ll < SEQ) {
            const float* hr = h1 + ((size_t)(b << 10 | ll)) * DIMC;
            s += hr[2*d]   * cw[(2*d)   * 3 + j];
            s += hr[2*d+1] * cw[(2*d+1) * 3 + j];
        }
    }
    u[(size_t)rowi * D_IN + d] = s / (1.f + __expf(-s));   // silu
}

// x_proj (48x128) + dt_proj (128x16) + softplus, per row
// grid: MROWS blocks, 128 threads
__global__ void xproj_kernel(const float* __restrict__ u,
                             const float* __restrict__ xw,   // [48][128]
                             const float* __restrict__ dtw,  // [128][16]
                             const float* __restrict__ dtb,  // [128]
                             float* __restrict__ delta,
                             float* __restrict__ Bm,
                             float* __restrict__ Cm) {
    __shared__ float us[D_IN];
    __shared__ float xd[DT_RANK + 2 * D_STATE];
    int rowi = blockIdx.x;
    int t = threadIdx.x;
    us[t] = u[(size_t)rowi * D_IN + t];
    __syncthreads();
    if (t < 48) {
        float s = 0.f;
        #pragma unroll 4
        for (int d = 0; d < D_IN; ++d) s += us[d] * xw[t * D_IN + d];
        xd[t] = s;
        if (t >= DT_RANK && t < DT_RANK + D_STATE)
            Bm[(size_t)rowi * D_STATE + (t - DT_RANK)] = s;
        else if (t >= DT_RANK + D_STATE)
            Cm[(size_t)rowi * D_STATE + (t - DT_RANK - D_STATE)] = s;
    }
    __syncthreads();
    float s = dtb[t];
    #pragma unroll
    for (int k = 0; k < DT_RANK; ++k) s += xd[k] * dtw[t * DT_RANK + k];
    // softplus
    delta[(size_t)rowi * D_IN + t] = (s > 20.f) ? s : log1pf(__expf(s));
}

// selective scan, one thread per (dir, b, d); 16 states in registers
// launch: 48 blocks x 128 threads = 6144 = 2 * 24 * 128
__global__ void scan_kernel(const float* __restrict__ u,
                            const float* __restrict__ delta,
                            const float* __restrict__ Bm,
                            const float* __restrict__ Cm,
                            const float* __restrict__ A_log,
                            const float* __restrict__ Ab_log,
                            float* __restrict__ yf,
                            float* __restrict__ yb) {
    int tid = blockIdx.x * blockDim.x + threadIdx.x;
    int dir = tid >= NSEQ * D_IN;
    int bd = tid - dir * NSEQ * D_IN;
    int b = bd >> 7, d = bd & 127;
    const float* Alog = dir ? Ab_log : A_log;
    float* yo = dir ? yb : yf;

    float A[D_STATE], h[D_STATE];
    #pragma unroll
    for (int n = 0; n < D_STATE; ++n) { A[n] = -__expf(Alog[d * D_STATE + n]); h[n] = 0.f; }

    for (int s = 0; s < SEQ; ++s) {
        int l = dir ? (SEQ - 1 - s) : s;
        size_t rowi = (size_t)(b << 10 | l);
        float dl = delta[rowi * D_IN + d];
        float uu = u[rowi * D_IN + d];
        float du = dl * uu;
        const float* Br = Bm + rowi * D_STATE;
        const float* Cr = Cm + rowi * D_STATE;
        float y = 0.f;
        #pragma unroll
        for (int n = 0; n < D_STATE; ++n) {
            h[n] = __expf(dl * A[n]) * h[n] + du * Br[n];
            y += h[n] * Cr[n];
        }
        yo[rowi * D_IN + d] = y;
    }
}

// cat = [ yf + yb + 2*D*u , z=u ]   (256 channels per row)
__global__ void cat_kernel(const float* __restrict__ yf,
                           const float* __restrict__ yb,
                           const float* __restrict__ u,
                           const float* __restrict__ Dv,
                           float* __restrict__ cat) {
    size_t idx = (size_t)blockIdx.x * blockDim.x + threadIdx.x;
    size_t rowi = idx >> 8;
    int c = (int)(idx & 255);
    float v;
    if (c < D_IN)
        v = yf[rowi * D_IN + c] + yb[rowi * D_IN + c] + 2.f * Dv[c] * u[rowi * D_IN + c];
    else
        v = u[rowi * D_IN + (c - D_IN)];
    cat[idx] = v;
}

// rmsnorm over last dim (256); one block per row
__global__ void rmsnorm_kernel(const float* __restrict__ x,
                               const float* __restrict__ w,
                               float* __restrict__ out) {
    __shared__ float red[DIMC];
    int rowi = blockIdx.x, t = threadIdx.x;
    float v = x[(size_t)rowi * DIMC + t];
    red[t] = v * v;
    __syncthreads();
    for (int s = DIMC / 2; s > 0; s >>= 1) {
        if (t < s) red[t] += red[t + s];
        __syncthreads();
    }
    float scale = rsqrtf(red[0] * (1.f / DIMC) + 1e-5f);
    out[(size_t)rowi * DIMC + t] = v * scale * w[t];
}

// depthwise 3x3 SAME conv over (F=3, t=1024) grid with 1024 channels + exact GeLU
// grid: (MROWS, 4), block 256
__global__ void conv3x3_gelu_kernel(const float* __restrict__ hb,
                                    const float* __restrict__ cw,  // [1024][1][3][3]
                                    const float* __restrict__ cb,
                                    float* __restrict__ gb) {
    int sp = blockIdx.x;                 // (b*3 + f)*1024 + tt
    int c = blockIdx.y * 256 + threadIdx.x;
    int b = sp / (3 * SEQ);
    int rem = sp - b * 3 * SEQ;
    int f = rem >> 10, tt = rem & 1023;
    float s = cb[c];
    #pragma unroll
    for (int df = -1; df <= 1; ++df) {
        int ff = f + df;
        if (ff < 0 || ff >= 3) continue;
        #pragma unroll
        for (int dt = -1; dt <= 1; ++dt) {
            int t2 = tt + dt;
            if (t2 < 0 || t2 >= SEQ) continue;
            s += hb[(((size_t)b * 3 + ff) * SEQ + t2) * HID + c] *
                 cw[c * 9 + (df + 1) * 3 + (dt + 1)];
        }
    }
    gb[(size_t)sp * HID + c] = 0.5f * s * (1.f + erff(s * 0.70710678118f));
}

// ---------------------------------------------------------------------------
extern "C" void kernel_launch(void* const* d_in, const int* in_sizes, int n_in,
                              void* d_out, int out_size, void* d_ws, size_t ws_size,
                              hipStream_t stream) {
    (void)in_sizes; (void)n_in; (void)out_size; (void)ws_size;
    const float* x_in   = (const float*)d_in[0];
    const float* norm_w = (const float*)d_in[31];

    // workspace carve-out (bytes)
    char* ws = (char*)d_ws;
    const size_t SZ_XC = (size_t)MROWS * DIMC * 4;   // 25.2 MB
    const size_t SZ_U  = (size_t)MROWS * D_IN * 4;   // 12.6 MB
    const size_t SZ_BC = (size_t)MROWS * D_STATE * 4;
    const size_t SZ_HB = (size_t)MROWS * HID * 4;    // 100.7 MB
    float* xcur  = (float*)(ws);
    float* h1    = (float*)(ws + SZ_XC);             // reused: h1 -> cat -> xn
    float* u     = (float*)(ws + 2 * SZ_XC);
    float* delta = (float*)(ws + 2 * SZ_XC + SZ_U);
    float* yf    = (float*)(ws + 2 * SZ_XC + 2 * SZ_U);
    float* yb    = (float*)(ws + 2 * SZ_XC + 3 * SZ_U);
    float* Bm    = (float*)(ws + 2 * SZ_XC + 4 * SZ_U);
    float* Cm    = (float*)(ws + 2 * SZ_XC + 4 * SZ_U + SZ_BC);
    float* hb    = (float*)(ws + 2 * SZ_XC + 4 * SZ_U + 2 * SZ_BC);
    float* gb    = (float*)(ws + 2 * SZ_XC + 4 * SZ_U + 2 * SZ_BC + SZ_HB);

    hipMemcpyAsync(xcur, x_in, SZ_XC, hipMemcpyDeviceToDevice, stream);

    const dim3 g256(MROWS / 128, DIMC / 32);     // (192, 8)
    const dim3 g1024(MROWS / 128, HID / 32);     // (192, 32)

    for (int layer = 0; layer < 2; ++layer) {
        const float* const* p = (const float* const*)(d_in + 1 + layer * 12);
        const float* ip_w = p[0],  *ip_b = p[1];
        const float* cw   = p[2],  *cb   = p[3];
        const float* Alog = p[4],  *Ablog= p[5];
        const float* Dv   = p[6];
        const float* xw   = p[7],  *dtw  = p[8], *dtb = p[9];
        const float* op_w = p[10], *op_b = p[11];

        gemm_wmma_kernel<<<g256, 128, 0, stream>>>(xcur, ip_w, ip_b, nullptr, h1, DIMC, DIMC);
        conv_silu_kernel<<<MROWS, 128, 0, stream>>>(h1, cw, cb, u);
        xproj_kernel<<<MROWS, 128, 0, stream>>>(u, xw, dtw, dtb, delta, Bm, Cm);
        scan_kernel<<<48, 128, 0, stream>>>(u, delta, Bm, Cm, Alog, Ablog, yf, yb);
        cat_kernel<<<(MROWS * DIMC) / 256, 256, 0, stream>>>(yf, yb, u, Dv, h1);
        gemm_wmma_kernel<<<g256, 128, 0, stream>>>(h1, op_w, op_b, xcur, xcur, DIMC, DIMC);
    }

    const float* fc1_w = (const float*)d_in[25];
    const float* fc1_b = (const float*)d_in[26];
    const float* mcw   = (const float*)d_in[27];
    const float* mcb   = (const float*)d_in[28];
    const float* fc2_w = (const float*)d_in[29];
    const float* fc2_b = (const float*)d_in[30];

    rmsnorm_kernel<<<MROWS, DIMC, 0, stream>>>(xcur, norm_w, h1);        // h1 = xn
    gemm_wmma_kernel<<<g1024, 128, 0, stream>>>(h1, fc1_w, fc1_b, nullptr, hb, HID, DIMC);
    conv3x3_gelu_kernel<<<dim3(MROWS, 4), 256, 0, stream>>>(hb, mcw, mcb, gb);
    gemm_wmma_kernel<<<g256, 128, 0, stream>>>(gb, fc2_w, fc2_b, xcur, xcur, DIMC, HID);
    rmsnorm_kernel<<<MROWS, DIMC, 0, stream>>>(xcur, norm_w, (float*)d_out);
}
